// ResNetCifar_80418967650604
// MI455X (gfx1250) — matrix operand order repaired and checked by reading, not compiled, not run on hardware
//
#include <hip/hip_runtime.h>
#include <hip/hip_bf16.h>

// Problem constants (match reference)
constexpr int B = 256;
constexpr int K = 1024;
constexpr int D = 2048;
constexpr int N = 50000;
constexpr float INV_T  = 1.0f / 0.07f;
constexpr float INV_Z  = 1.0f / 52580.0f;
constexpr float MOM    = 0.5f;

typedef __attribute__((ext_vector_type(16))) _Float16 v16h;
typedef __attribute__((ext_vector_type(8)))  float    v8f;
typedef __attribute__((ext_vector_type(4)))  float    vf4;

// ---------------------------------------------------------------------------
// Kernel 1: features = x / ||x||  (one block per row, 256 threads, 8 elem/thr)
// ---------------------------------------------------------------------------
__global__ __launch_bounds__(256) void feat_normalize(const float* __restrict__ x,
                                                      float* __restrict__ feat) {
    __shared__ float red[256];
    const int b   = blockIdx.x;
    const int tid = threadIdx.x;
    const float* xr = x + (size_t)b * D;
    float v[8];
    float s = 0.0f;
#pragma unroll
    for (int i = 0; i < 8; ++i) {
        v[i] = xr[tid + 256 * i];
        s += v[i] * v[i];
    }
    red[tid] = s;
    __syncthreads();
    for (int off = 128; off > 0; off >>= 1) {
        if (tid < off) red[tid] += red[tid + off];
        __syncthreads();
    }
    const float inv = 1.0f / sqrtf(red[0]);
    float* fr = feat + (size_t)b * D;
#pragma unroll
    for (int i = 0; i < 8; ++i) fr[tid + 256 * i] = v[i] * inv;
}

// ---------------------------------------------------------------------------
// Kernel 2: temp = memory  (410MB clone). Regular-temporal loads keep the
// memory bank warming L2 for the gather kernel; non-temporal stores keep the
// write-once clone from evicting it.
// ---------------------------------------------------------------------------
__global__ __launch_bounds__(256) void copy_bank(const vf4* __restrict__ src,
                                                 vf4* __restrict__ dst,
                                                 long n4) {
    long i      = ((long)blockIdx.x * blockDim.x + threadIdx.x) * 2;
    long stride = (long)gridDim.x * blockDim.x * 2;
    for (; i < n4; i += stride) {           // n4 is even
        vf4 a = src[i];
        vf4 b = src[i + 1];
        __builtin_nontemporal_store(a, &dst[i]);
        __builtin_nontemporal_store(b, &dst[i + 1]);
    }
}

// ---------------------------------------------------------------------------
// Kernel 3: temp[y[b]] = normalize(memory[y[b]]*MOM + feat[b]*(1-MOM))
// (reads pre-update memory; one block per b; runs after the clone)
// ---------------------------------------------------------------------------
__global__ __launch_bounds__(256) void update_positive(const float* __restrict__ memory,
                                                       const float* __restrict__ feat,
                                                       const int* __restrict__ y,
                                                       float* __restrict__ temp) {
    __shared__ float red[256];
    const int b   = blockIdx.x;
    const int tid = threadIdx.x;
    const int row = y[b];
    const float* mr = memory + (size_t)row * D;
    const float* fr = feat   + (size_t)b   * D;
    float v[8];
    float s = 0.0f;
#pragma unroll
    for (int i = 0; i < 8; ++i) {
        const int j = tid + 256 * i;
        v[i] = mr[j] * MOM + fr[j] * (1.0f - MOM);
        s += v[i] * v[i];
    }
    red[tid] = s;
    __syncthreads();
    for (int off = 128; off > 0; off >>= 1) {
        if (tid < off) red[tid] += red[tid + off];
        __syncthreads();
    }
    const float inv = 1.0f / sqrtf(red[0]);
    float* tr = temp + (size_t)row * D;
#pragma unroll
    for (int i = 0; i < 8; ++i) tr[tid + 256 * i] = v[i] * inv;
}

// ---------------------------------------------------------------------------
// Kernel 4: out[b,k] = exp(dot(memory[idx'[b,k]], feat[b]) / T) / Z
// Split-f16 error-compensated WMMA: dot = ah*bh + ah*bl + al*bh (f32 acc).
// grid = B*8 blocks x 256 threads; each wave owns one 16-row k-tile
// (16384 waves total -> deep memory-level parallelism for the gather).
// ---------------------------------------------------------------------------
__global__ __launch_bounds__(256) void nce_dot_wmma(const float* __restrict__ memory,
                                                    const float* __restrict__ feat,
                                                    const int* __restrict__ idx,
                                                    const int* __restrict__ y,
                                                    float* __restrict__ out) {
    __shared__ __align__(32) _Float16 sfh[D];   // 4KB: f16 hi part of feat[b]
    __shared__ __align__(32) _Float16 sfl[D];   // 4KB: f16 lo residual
    const int b   = blockIdx.x >> 3;
    const int tg  = blockIdx.x & 7;
    const int tid = threadIdx.x;

    // stage features for this b into LDS as hi/lo f16 pair
    for (int i = tid; i < D; i += 256) {
        const float f = feat[(size_t)b * D + i];
        const _Float16 h = (_Float16)f;
        sfh[i] = h;
        sfl[i] = (_Float16)(f - (float)h);
    }
    __syncthreads();

    const int wave  = tid >> 5;       // 0..7
    const int lane  = tid & 31;       // wave32
    const int m     = lane & 15;      // matrix row owned by this lane
    const bool hi   = lane >= 16;
    const int alo   = hi ? 8  : 0;    // A fragment: K-offset group per half-wave
    const int bbase = hi ? 8 : 0;     // B fragment: K-offset group per half-wave (same as A)

    const int t = tg * 8 + wave;      // this wave's k-tile (0..63)
    const int k = t * 16 + m;
    const int r = (k == 0) ? y[b] : idx[b * K + k];
    const float* arow = memory + (size_t)r * D;

    v8f acc = {};
    for (int d = 0; d < D; d += 32) {
        // ---- gather 16 f32 of this lane's A row (16-bit A 16x32 layout):
        // lane<16 : K = d+{0..7} in h0..7,  d+{16..23} in h8..15
        // lane>=16: K = d+{8..15} in h0..7, d+{24..31} in h8..15
        const vf4* p0 = (const vf4*)(arow + d + alo);
        const vf4* p1 = (const vf4*)(arow + d + 16 + alo);
        const vf4 q0 = p0[0], q1 = p0[1];
        const vf4 q2 = p1[0], q3 = p1[1];
        float fl[16];
#pragma unroll
        for (int j = 0; j < 4; ++j) {
            fl[j]      = q0[j];
            fl[4 + j]  = q1[j];
            fl[8 + j]  = q2[j];
            fl[12 + j] = q3[j];
        }
        // split each f32 into hi f16 + lo f16 residual
        v16h ah, al;
#pragma unroll
        for (int j = 0; j < 16; ++j) {
            const _Float16 h = (_Float16)fl[j];
            ah[j] = h;
            al[j] = (_Float16)(fl[j] - (float)h);
        }
        // ---- B fragments (32x16): same K order as A (lane<16: d+{0..7,16..23},
        // lane>=16: d+{8..15,24..31}); all 16 columns identical = feature values.
        v16h bh, bl;
#pragma unroll
        for (int j = 0; j < 16; ++j) {
            const int kk = d + bbase + (j < 8 ? j : 8 + j);
            bh[j] = sfh[kk];
            bl[j] = sfl[kk];
        }

        acc = __builtin_amdgcn_wmma_f32_16x16x32_f16(
            false, ah, false, bh, (short)0, acc, false, false);
        acc = __builtin_amdgcn_wmma_f32_16x16x32_f16(
            false, ah, false, bl, (short)0, acc, false, false);
        acc = __builtin_amdgcn_wmma_f32_16x16x32_f16(
            false, al, false, bh, (short)0, acc, false, false);
    }

    // D matrix: lane 0 (N=0) holds M=0..7 in acc[0..7]; lane 16 holds M=8..15
    if (m == 0) {
        const int mbase = hi ? 8 : 0;
#pragma unroll
        for (int vv = 0; vv < 8; ++vv) {
            out[(size_t)b * K + t * 16 + mbase + vv] =
                __expf(acc[vv] * INV_T) * INV_Z;
        }
    }
}

// ---------------------------------------------------------------------------
extern "C" void kernel_launch(void* const* d_in, const int* in_sizes, int n_in,
                              void* d_out, int out_size, void* d_ws, size_t ws_size,
                              hipStream_t stream) {
    const float* x      = (const float*)d_in[0];   // [B, D]
    const int*   y      = (const int*)  d_in[1];   // [B]
    const int*   idx    = (const int*)  d_in[2];   // [B, K]
    const float* memory = (const float*)d_in[3];   // [N, D]

    float* out  = (float*)d_out;                   // [B, K]
    float* temp = out + (size_t)B * K;             // [N, D]
    float* feat = (float*)d_ws;                    // [B, D] scratch (2 MB)

    // 1. normalize features
    feat_normalize<<<B, 256, 0, stream>>>(x, feat);

    // 2. clone memory bank into temp (NT stores preserve L2 for the gathers)
    const long n4 = (long)N * D / 4;
    copy_bank<<<8192, 256, 0, stream>>>((const vf4*)memory, (vf4*)temp, n4);

    // 3. momentum-update + renormalize positive rows (after clone, same stream)
    update_positive<<<B, 256, 0, stream>>>(memory, feat, y, temp);

    // 4. gathered batched GEMV via split-f16 compensated WMMA
    nce_dot_wmma<<<B * 8, 256, 0, stream>>>(memory, feat, idx, y, out);
}
